// Def_Bottle_S_55276229099716
// MI455X (gfx1250) — matrix operand order repaired and checked by reading, not compiled, not run on hardware
//
#include <hip/hip_runtime.h>
#include <cstdint>

// ---------------------------------------------------------------------------
// CDNA5 (gfx1250) deformable bottleneck, bf16 WMMA with f32 accumulation.
// ---------------------------------------------------------------------------

typedef __bf16 bf16_t;
typedef __attribute__((ext_vector_type(16))) __bf16 v16bf;
typedef __attribute__((ext_vector_type(8)))  __bf16 v8bf;
typedef __attribute__((ext_vector_type(8)))  float  v8f;
typedef __attribute__((ext_vector_type(4)))  float  v4f;

static constexpr int kB   = 16;    // batch
static constexpr int kCin = 1024;  // input channels
static constexpr int kC   = 256;   // mid channels
static constexpr int kC4  = 1024;  // output channels
static constexpr int kHW  = 1024;  // 32*32 spatial
static constexpr int kK2  = 2304;  // 9*256 deform K

// ---- WMMA fragment helpers (layouts per CDNA5 ISA 7.12.2) ------------------

// A (16x32 bf16, row-major source, one row per lane m=lane&15):
// VGPR0-3 hold K = 8*half + {0..7}; VGPR4-7 hold K = 16 + 8*half + {0..7}.
__device__ __forceinline__ v16bf load_frag_a(const bf16_t* rowp, int lh) {
  v16bf f;
  *reinterpret_cast<v8bf*>(&f)       = *reinterpret_cast<const v8bf*>(rowp + 8 * lh);
  *(reinterpret_cast<v8bf*>(&f) + 1) = *reinterpret_cast<const v8bf*>(rowp + 16 + 8 * lh);
  return f;
}

// B (32x16 bf16, K-contiguous per column n=lane&15):
// VGPR0-3 hold K = 8*half + {0..7}; VGPR4-7 hold K = 16 + 8*half + {0..7}.
__device__ __forceinline__ v16bf load_frag_b(const bf16_t* colp, int lh) {
  v16bf f;
  const v8bf* p = reinterpret_cast<const v8bf*>(colp);
  *reinterpret_cast<v8bf*>(&f)       = p[lh];
  *(reinterpret_cast<v8bf*>(&f) + 1) = p[2 + lh];
  return f;
}

__device__ __forceinline__ v8f wmma_bf16(v16bf a, v16bf b, v8f c) {
  return __builtin_amdgcn_wmma_f32_16x16x32_bf16(false, a, false, b, (short)0, c,
                                                 false, false);
}

// ---------------------------------------------------------------------------
// Kernel 0: weight conversion / reordering + BN folding.
// offw/w2 K-index is tap-major: r = k*256 + c  (k = ky*3+kx).
// ---------------------------------------------------------------------------
__global__ void prep_kernel(
    const float* __restrict__ conv1_w, const float* __restrict__ conv1_b,
    const float* __restrict__ g1, const float* __restrict__ b1,
    const float* __restrict__ m1, const float* __restrict__ v1,
    const float* __restrict__ off_w,
    const float* __restrict__ dconv_w, const float* __restrict__ dconv_b,
    const float* __restrict__ g2, const float* __restrict__ b2,
    const float* __restrict__ m2, const float* __restrict__ v2,
    const float* __restrict__ conv2_w, const float* __restrict__ conv2_b,
    const float* __restrict__ g3, const float* __restrict__ b3,
    const float* __restrict__ m3, const float* __restrict__ v3,
    bf16_t* __restrict__ w1h, bf16_t* __restrict__ offwh,
    bf16_t* __restrict__ w2h, bf16_t* __restrict__ w3h,
    float* __restrict__ scale1, float* __restrict__ bias1,
    float* __restrict__ scale2, float* __restrict__ bias2,
    float* __restrict__ scale3, float* __restrict__ bias3) {
  const int tid    = blockIdx.x * blockDim.x + threadIdx.x;
  const int stride = gridDim.x * blockDim.x;

  for (int i = tid; i < kC * kCin; i += stride) w1h[i] = (bf16_t)conv1_w[i];

  for (int i = tid; i < 32 * kK2; i += stride) {
    int o = i / kK2, r = i % kK2;
    int k = r >> 8, c = r & 255;
    // off_w shape (18, 256, 3, 3): flat = (o*256 + c)*9 + k
    offwh[i] = (o < 18) ? (bf16_t)off_w[(o * 256 + c) * 9 + k] : (bf16_t)0.0f;
  }
  for (int i = tid; i < kC * kK2; i += stride) {
    int o = i / kK2, r = i % kK2;
    int k = r >> 8, c = r & 255;
    w2h[i] = (bf16_t)dconv_w[(o * 256 + c) * 9 + k];
  }
  for (int i = tid; i < kC4 * kC; i += stride) w3h[i] = (bf16_t)conv2_w[i];

  for (int i = tid; i < kC; i += stride) {
    float s1 = g1[i] * rsqrtf(v1[i] + 1e-5f);
    scale1[i] = s1;
    bias1[i]  = (conv1_b[i] - m1[i]) * s1 + b1[i];
    float s2 = g2[i] * rsqrtf(v2[i] + 1e-5f);
    scale2[i] = s2;
    bias2[i]  = (dconv_b[i] - m2[i]) * s2 + b2[i];
  }
  for (int i = tid; i < kC4; i += stride) {
    float s3 = g3[i] * rsqrtf(v3[i] + 1e-5f);
    scale3[i] = s3;
    bias3[i]  = (conv2_b[i] - m3[i]) * s3 + b3[i];
  }
}

// ---------------------------------------------------------------------------
// Kernel 1: x (B,Cin,HW) f32  ->  xT (B,HW,Cin) bf16  (WMMA B operand layout)
// ---------------------------------------------------------------------------
__global__ void xpose_kernel(const float* __restrict__ x, bf16_t* __restrict__ xT) {
  int64_t i = (int64_t)blockIdx.x * blockDim.x + threadIdx.x;
  if (i >= (int64_t)kB * kCin * kHW) return;
  int n  = (int)(i % kHW);
  int64_t t = i / kHW;
  int ci = (int)(t % kCin);
  int b  = (int)(t / kCin);
  xT[((int64_t)b * kHW + n) * kCin + ci] = (bf16_t)x[i];
}

// ---------------------------------------------------------------------------
// Kernel 2: GEMM1  out1[o][n] = W1(256x1024) * xT[b] (1024x1024), BN1 + ReLU.
// Wave tile 64x32, workgroup (8 waves) tiles 256x64. Output stored [n][c] in
// both f32 (bilinear source) and bf16 (GEMM operand).
// ---------------------------------------------------------------------------
__global__ __launch_bounds__(256) void gemm1_kernel(
    const bf16_t* __restrict__ w1, const bf16_t* __restrict__ xT,
    const float* __restrict__ scale1, const float* __restrict__ bias1,
    float* __restrict__ out1f, bf16_t* __restrict__ out1h) {
  const int lane = threadIdx.x & 31;
  const int wave = threadIdx.x >> 5;
  const int b    = blockIdx.y;
  const int m0   = (wave & 3) * 64;
  const int n0   = blockIdx.x * 64 + (wave >> 2) * 32;
  const int lm   = lane & 15, lh = lane >> 4;

  v8f acc[4][2];
#pragma unroll
  for (int mi = 0; mi < 4; ++mi)
#pragma unroll
    for (int ni = 0; ni < 2; ++ni)
      acc[mi][ni] = v8f{0.f, 0.f, 0.f, 0.f, 0.f, 0.f, 0.f, 0.f};

  const bf16_t* bbase = xT + (int64_t)b * kHW * kCin;
  for (int kt = 0; kt < kCin; kt += 32) {
    v16bf afr[4], bfr[2];
#pragma unroll
    for (int mi = 0; mi < 4; ++mi) {
      int row = m0 + mi * 16 + lm;
      afr[mi] = load_frag_a(w1 + (int64_t)row * kCin + kt, lh);
    }
#pragma unroll
    for (int ni = 0; ni < 2; ++ni) {
      int col = n0 + ni * 16 + lm;
      const bf16_t* bp = bbase + (int64_t)col * kCin + kt;
      __builtin_prefetch(bp + 32, 0, 0);  // global_prefetch_b8: next K chunk
      bfr[ni] = load_frag_b(bp, lh);
    }
#pragma unroll
    for (int mi = 0; mi < 4; ++mi)
#pragma unroll
      for (int ni = 0; ni < 2; ++ni)
        acc[mi][ni] = wmma_bf16(afr[mi], bfr[ni], acc[mi][ni]);
  }

#pragma unroll
  for (int mi = 0; mi < 4; ++mi)
#pragma unroll
    for (int ni = 0; ni < 2; ++ni) {
      int n  = n0 + ni * 16 + lm;
      int mb = m0 + mi * 16 + lh * 8;
      int64_t base = ((int64_t)b * kHW + n) * kC;
      float y[8];
      v8bf yb;
#pragma unroll
      for (int v = 0; v < 8; ++v) {
        int m  = mb + v;
        float t = acc[mi][ni][v] * scale1[m] + bias1[m];
        t = fmaxf(t, 0.0f);
        y[v]  = t;
        yb[v] = (bf16_t)t;
      }
      *reinterpret_cast<v4f*>(&out1f[base + mb])     = *reinterpret_cast<v4f*>(&y[0]);
      *reinterpret_cast<v4f*>(&out1f[base + mb + 4]) = *reinterpret_cast<v4f*>(&y[4]);
      *reinterpret_cast<v8bf*>(&out1h[base + mb])    = yb;
    }
}

// ---------------------------------------------------------------------------
// Kernel 3: offset conv3x3 as WMMA GEMM. M=32 (18 padded), K tap-major
// (k*256+c): B fragment for tap k is a spatially shifted contiguous load of
// out1h[b][ns][c], zeroed at borders per lane-column. One 16-col tile/wave.
// ---------------------------------------------------------------------------
__global__ __launch_bounds__(256) void offconv_kernel(
    const bf16_t* __restrict__ offw, const bf16_t* __restrict__ out1h,
    const float* __restrict__ off_b, float* __restrict__ offset) {
  const int lane = threadIdx.x & 31;
  const int wave = threadIdx.x >> 5;
  const int b    = blockIdx.y;
  const int n0   = (blockIdx.x * 8 + wave) * 16;
  const int lm   = lane & 15, lh = lane >> 4;

  const int n = n0 + lm;
  const int h = n >> 5, w = n & 31;

  v8f acc[2];
  acc[0] = v8f{0.f, 0.f, 0.f, 0.f, 0.f, 0.f, 0.f, 0.f};
  acc[1] = acc[0];

  for (int k = 0; k < 9; ++k) {
    const int dy = k / 3 - 1, dx = k % 3 - 1;
    const int hs = h + dy, ws2 = w + dx;
    const bool valid = (hs >= 0) && (hs < 32) && (ws2 >= 0) && (ws2 < 32);
    const int ns = hs * 32 + ws2;
    const bf16_t* colp = out1h + ((int64_t)b * kHW + ns) * kC;
#pragma unroll
    for (int kc = 0; kc < kC; kc += 32) {
      v16bf bfr;
      if (valid) {
        bfr = load_frag_b(colp + kc, lh);
      } else {
#pragma unroll
        for (int e = 0; e < 16; ++e) bfr[e] = (bf16_t)0.0f;
      }
      v16bf a0 = load_frag_a(offw + (int64_t)(0 + lm) * kK2 + k * 256 + kc, lh);
      v16bf a1 = load_frag_a(offw + (int64_t)(16 + lm) * kK2 + k * 256 + kc, lh);
      acc[0] = wmma_bf16(a0, bfr, acc[0]);
      acc[1] = wmma_bf16(a1, bfr, acc[1]);
    }
  }

#pragma unroll
  for (int mi = 0; mi < 2; ++mi)
#pragma unroll
    for (int v = 0; v < 8; ++v) {
      int m = mi * 16 + lh * 8 + v;
      if (m < 18)
        offset[((int64_t)b * 18 + m) * kHW + n] = acc[mi][v] + off_b[m];
    }
}

// ---------------------------------------------------------------------------
// Kernel 4: bilinear sampling -> deform im2col val[b][n][k*256+c] (bf16).
// One wave per (b, k, n); lanes stride channels (coalesced ld/st).
// ---------------------------------------------------------------------------
__global__ void sample_kernel(const float* __restrict__ out1f,
                              const float* __restrict__ offset,
                              bf16_t* __restrict__ val) {
  const int lane = threadIdx.x;                       // 0..31
  const int n    = blockIdx.x * blockDim.y + threadIdx.y;
  const int k    = blockIdx.y;
  const int b    = blockIdx.z;
  const int h = n >> 5, w = n & 31;

  const float dy = offset[((int64_t)b * 18 + 2 * k) * kHW + n];
  const float dx = offset[((int64_t)b * 18 + 2 * k + 1) * kHW + n];
  const float py = (float)h + (float)(k / 3 - 1) + dy;
  const float px = (float)w + (float)(k % 3 - 1) + dx;
  const float y0f = floorf(py), x0f = floorf(px);
  const float ly = py - y0f, lx = px - x0f;
  const int y0 = (int)y0f, x0 = (int)x0f;
  const int y1 = y0 + 1, x1 = x0 + 1;
  const float vy0 = (y0 >= 0 && y0 < 32) ? 1.f : 0.f;
  const float vy1 = (y1 >= 0 && y1 < 32) ? 1.f : 0.f;
  const float vx0 = (x0 >= 0 && x0 < 32) ? 1.f : 0.f;
  const float vx1 = (x1 >= 0 && x1 < 32) ? 1.f : 0.f;
  const int cy0 = min(max(y0, 0), 31), cy1 = min(max(y1, 0), 31);
  const int cx0 = min(max(x0, 0), 31), cx1 = min(max(x1, 0), 31);
  const int64_t bb = (int64_t)b * kHW;
  const int64_t p00 = (bb + (cy0 * 32 + cx0)) * kC;
  const int64_t p01 = (bb + (cy0 * 32 + cx1)) * kC;
  const int64_t p10 = (bb + (cy1 * 32 + cx0)) * kC;
  const int64_t p11 = (bb + (cy1 * 32 + cx1)) * kC;
  const float w00 = (1.f - ly) * (1.f - lx) * vy0 * vx0;
  const float w01 = (1.f - ly) * lx * vy0 * vx1;
  const float w10 = ly * (1.f - lx) * vy1 * vx0;
  const float w11 = ly * lx * vy1 * vx1;
  const int64_t obase = (bb + n) * kK2 + k * 256;

  for (int c = lane; c < kC; c += 32) {
    float s = w00 * out1f[p00 + c] + w01 * out1f[p01 + c] +
              w10 * out1f[p10 + c] + w11 * out1f[p11 + c];
    val[obase + c] = (bf16_t)s;
  }
}

// ---------------------------------------------------------------------------
// Kernel 5: deform GEMM  out2[o][n] = W2(256x2304) * val[b] (2304x1024), BN2.
// Same 64x32 wave tile as GEMM1; output bf16 [n][c] for conv2's B operand.
// ---------------------------------------------------------------------------
__global__ __launch_bounds__(256) void gemm2_kernel(
    const bf16_t* __restrict__ w2, const bf16_t* __restrict__ val,
    const float* __restrict__ scale2, const float* __restrict__ bias2,
    bf16_t* __restrict__ out2h) {
  const int lane = threadIdx.x & 31;
  const int wave = threadIdx.x >> 5;
  const int b    = blockIdx.y;
  const int m0   = (wave & 3) * 64;
  const int n0   = blockIdx.x * 64 + (wave >> 2) * 32;
  const int lm   = lane & 15, lh = lane >> 4;

  v8f acc[4][2];
#pragma unroll
  for (int mi = 0; mi < 4; ++mi)
#pragma unroll
    for (int ni = 0; ni < 2; ++ni)
      acc[mi][ni] = v8f{0.f, 0.f, 0.f, 0.f, 0.f, 0.f, 0.f, 0.f};

  const bf16_t* bbase = val + (int64_t)b * kHW * kK2;
  for (int kt = 0; kt < kK2; kt += 32) {
    v16bf afr[4], bfr[2];
#pragma unroll
    for (int mi = 0; mi < 4; ++mi) {
      int row = m0 + mi * 16 + lm;
      afr[mi] = load_frag_a(w2 + (int64_t)row * kK2 + kt, lh);
    }
#pragma unroll
    for (int ni = 0; ni < 2; ++ni) {
      int col = n0 + ni * 16 + lm;
      const bf16_t* bp = bbase + (int64_t)col * kK2 + kt;
      __builtin_prefetch(bp + 32, 0, 0);
      bfr[ni] = load_frag_b(bp, lh);
    }
#pragma unroll
    for (int mi = 0; mi < 4; ++mi)
#pragma unroll
      for (int ni = 0; ni < 2; ++ni)
        acc[mi][ni] = wmma_bf16(afr[mi], bfr[ni], acc[mi][ni]);
  }

#pragma unroll
  for (int mi = 0; mi < 4; ++mi)
#pragma unroll
    for (int ni = 0; ni < 2; ++ni) {
      int n  = n0 + ni * 16 + lm;
      int mb = m0 + mi * 16 + lh * 8;
      int64_t base = ((int64_t)b * kHW + n) * kC;
      v8bf yb;
#pragma unroll
      for (int v = 0; v < 8; ++v) {
        int m = mb + v;
        yb[v] = (bf16_t)(acc[mi][ni][v] * scale2[m] + bias2[m]);  // no ReLU
      }
      *reinterpret_cast<v8bf*>(&out2h[base + mb]) = yb;
    }
}

// ---------------------------------------------------------------------------
// Kernel 6: GEMM3  out[o][n] = W3(1024x256) * out2[b] (256x1024), BN3 + ReLU,
// f32 result in reference layout (B, 1024, 32, 32).
// ---------------------------------------------------------------------------
__global__ __launch_bounds__(256) void gemm3_kernel(
    const bf16_t* __restrict__ w3, const bf16_t* __restrict__ out2h,
    const float* __restrict__ scale3, const float* __restrict__ bias3,
    float* __restrict__ out) {
  const int lane = threadIdx.x & 31;
  const int wave = threadIdx.x >> 5;
  const int b    = blockIdx.z;
  const int m0   = blockIdx.y * 256 + (wave & 3) * 64;
  const int n0   = blockIdx.x * 64 + (wave >> 2) * 32;
  const int lm   = lane & 15, lh = lane >> 4;

  v8f acc[4][2];
#pragma unroll
  for (int mi = 0; mi < 4; ++mi)
#pragma unroll
    for (int ni = 0; ni < 2; ++ni)
      acc[mi][ni] = v8f{0.f, 0.f, 0.f, 0.f, 0.f, 0.f, 0.f, 0.f};

  const bf16_t* bbase = out2h + (int64_t)b * kHW * kC;
#pragma unroll
  for (int kt = 0; kt < kC; kt += 32) {
    v16bf afr[4], bfr[2];
#pragma unroll
    for (int mi = 0; mi < 4; ++mi) {
      int row = m0 + mi * 16 + lm;
      afr[mi] = load_frag_a(w3 + (int64_t)row * kC + kt, lh);
    }
#pragma unroll
    for (int ni = 0; ni < 2; ++ni) {
      int col = n0 + ni * 16 + lm;
      bfr[ni] = load_frag_b(bbase + (int64_t)col * kC + kt, lh);
    }
#pragma unroll
    for (int mi = 0; mi < 4; ++mi)
#pragma unroll
      for (int ni = 0; ni < 2; ++ni)
        acc[mi][ni] = wmma_bf16(afr[mi], bfr[ni], acc[mi][ni]);
  }

#pragma unroll
  for (int mi = 0; mi < 4; ++mi)
#pragma unroll
    for (int ni = 0; ni < 2; ++ni) {
      int n  = n0 + ni * 16 + lm;
      int mb = m0 + mi * 16 + lh * 8;
#pragma unroll
      for (int v = 0; v < 8; ++v) {
        int m = mb + v;
        float t = fmaxf(acc[mi][ni][v] * scale3[m] + bias3[m], 0.0f);
        out[((int64_t)b * kC4 + m) * kHW + n] = t;
      }
    }
}

// ---------------------------------------------------------------------------
extern "C" void kernel_launch(void* const* d_in, const int* in_sizes, int n_in,
                              void* d_out, int out_size, void* d_ws, size_t ws_size,
                              hipStream_t stream) {
  (void)in_sizes; (void)n_in; (void)out_size; (void)ws_size;
  const float* x       = (const float*)d_in[0];
  const float* conv1_w = (const float*)d_in[1];
  const float* conv1_b = (const float*)d_in[2];
  const float* bn1_g   = (const float*)d_in[3];
  const float* bn1_b   = (const float*)d_in[4];
  const float* bn1_m   = (const float*)d_in[5];
  const float* bn1_v   = (const float*)d_in[6];
  const float* off_w   = (const float*)d_in[7];
  const float* off_b   = (const float*)d_in[8];
  const float* dconv_w = (const float*)d_in[9];
  const float* dconv_b = (const float*)d_in[10];
  const float* bn2_g   = (const float*)d_in[11];
  const float* bn2_b   = (const float*)d_in[12];
  const float* bn2_m   = (const float*)d_in[13];
  const float* bn2_v   = (const float*)d_in[14];
  const float* conv2_w = (const float*)d_in[15];
  const float* conv2_b = (const float*)d_in[16];
  const float* bn3_g   = (const float*)d_in[17];
  const float* bn3_b   = (const float*)d_in[18];
  const float* bn3_m   = (const float*)d_in[19];
  const float* bn3_v   = (const float*)d_in[20];
  float* out = (float*)d_out;

  char* wp = (char*)d_ws;
  auto take = [&](size_t bytes) {
    char* p = wp;
    wp += (bytes + 255) & ~(size_t)255;
    return p;
  };
  bf16_t* xT     = (bf16_t*)take((size_t)kB * kHW * kCin * 2);   // 32 MB
  bf16_t* w1h    = (bf16_t*)take((size_t)kC * kCin * 2);
  bf16_t* offwh  = (bf16_t*)take((size_t)32 * kK2 * 2);
  bf16_t* w2h    = (bf16_t*)take((size_t)kC * kK2 * 2);
  bf16_t* w3h    = (bf16_t*)take((size_t)kC4 * kC * 2);
  float*  scale1 = (float*)take(kC * 4);
  float*  bias1  = (float*)take(kC * 4);
  float*  scale2 = (float*)take(kC * 4);
  float*  bias2  = (float*)take(kC * 4);
  float*  scale3 = (float*)take(kC4 * 4);
  float*  bias3  = (float*)take(kC4 * 4);
  float*  out1f  = (float*)take((size_t)kB * kHW * kC * 4);      // 16 MB
  bf16_t* out1h  = (bf16_t*)take((size_t)kB * kHW * kC * 2);     // 8 MB
  float*  offset = (float*)take((size_t)kB * 18 * kHW * 4);      // 1.1 MB
  bf16_t* val    = (bf16_t*)take((size_t)kB * kHW * kK2 * 2);    // 72 MB
  bf16_t* out2h  = (bf16_t*)take((size_t)kB * kHW * kC * 2);     // 8 MB

  prep_kernel<<<256, 256, 0, stream>>>(
      conv1_w, conv1_b, bn1_g, bn1_b, bn1_m, bn1_v, off_w, dconv_w, dconv_b,
      bn2_g, bn2_b, bn2_m, bn2_v, conv2_w, conv2_b, bn3_g, bn3_b, bn3_m, bn3_v,
      w1h, offwh, w2h, w3h, scale1, bias1, scale2, bias2, scale3, bias3);

  {
    int64_t total = (int64_t)kB * kCin * kHW;
    xpose_kernel<<<(unsigned)((total + 255) / 256), 256, 0, stream>>>(x, xT);
  }

  gemm1_kernel<<<dim3(kHW / 64, kB), 256, 0, stream>>>(w1h, xT, scale1, bias1,
                                                       out1f, out1h);

  offconv_kernel<<<dim3(kHW / 128, kB), 256, 0, stream>>>(offwh, out1h, off_b,
                                                          offset);

  sample_kernel<<<dim3(kHW / 8, 9, kB), dim3(32, 8), 0, stream>>>(out1f, offset,
                                                                  val);

  gemm2_kernel<<<dim3(kHW / 64, kB), 256, 0, stream>>>(w2h, val, scale2, bias2,
                                                       out2h);

  gemm3_kernel<<<dim3(kHW / 64, kC4 / 256, kB), 256, 0, stream>>>(
      w3h, out2h, scale3, bias3, out);
}
